// DynamicMoELayer_35777077576478
// MI455X (gfx1250) — compile-verified, matrix operand order checked by reading, not run on hardware
//
#include <hip/hip_runtime.h>
#include <hip/hip_bf16.h>

// ---- problem constants (match reference) ----
#define BB   4
#define TT   2048
#define BT   (BB*TT)     // 8192 tokens
#define DD   2048        // input dim
#define HH   4096        // router hidden
#define EE   16          // experts
#define OO   2048        // expert out dim
#define TAU  0.5f

typedef __attribute__((ext_vector_type(2)))  float          v2f;
typedef __attribute__((ext_vector_type(8)))  float          v8f;
typedef __attribute__((ext_vector_type(16))) __bf16         v16bf;
typedef __attribute__((ext_vector_type(4)))  unsigned int   v4u;
typedef __attribute__((ext_vector_type(4)))  int            v4i;

union BfPack { v4u q[2]; v16bf v; };

#if __has_builtin(__builtin_amdgcn_global_load_async_to_lds_b128) && \
    __has_builtin(__builtin_amdgcn_s_wait_asynccnt)
#define USE_ASYNC 1
typedef __attribute__((address_space(1))) v4i* gv4i_p;   // global int4*
typedef __attribute__((address_space(3))) v4i* lv4i_p;   // LDS int4*
#else
#define USE_ASYNC 0
#endif

__device__ __forceinline__ unsigned short f2bf(float f) {
    unsigned int u = __float_as_uint(f);
    unsigned int r = u + 0x7FFFu + ((u >> 16) & 1u);   // RNE
    return (unsigned short)(r >> 16);
}

// =====================================================================
// Kernel 1: fused router.  One block = 16 tokens, 8 waves.
//   h = relu(x@W1 + b1); logits = h@W2 + b2; softmax; threshold; gates, k.
// GEMM1 in fp32 WMMA 16x16x4; GEMM2 fused per 128-wide H chunk.
// LDS rows padded so half-wave operand fetches are bank-conflict-free.
// =====================================================================
__global__ __launch_bounds__(256) void router_kernel(
    const float* __restrict__ x,  const float* __restrict__ W1,
    const float* __restrict__ b1, const float* __restrict__ W2,
    const float* __restrict__ b2,
    float* __restrict__ gates,    // [BT][EE] in workspace
    float* __restrict__ kout)     // [BT] appended to d_out (stored as float)
{
    __shared__ __align__(16) float sX[16][68];     // x tile (+4 pad)
    __shared__ __align__(16) float sW1[128][66];   // W1^T tile (+2 pad)
    __shared__ __align__(16) float sW2[16][130];   // W2^T chunk (+2 pad)
    __shared__ __align__(16) float sH[8][16][18];  // per-wave h tile (+2 pad)
    __shared__ __align__(16) float sLogits[16][16];

    const int tid  = threadIdx.x;
    const int wave = tid >> 5;
    const int lane = tid & 31;
    const int m    = lane & 15;     // A row / B col / C col index
    const int g    = lane >> 4;     // lane half
    const int tok0 = blockIdx.x * 16;

    sLogits[tid >> 4][tid & 15] = 0.0f;

    v8f logit_acc = {0.f,0.f,0.f,0.f,0.f,0.f,0.f,0.f};

    for (int j = 0; j < HH / 128; ++j) {          // 32 chunks of 128 h-cols
        __syncthreads();                          // protect sW2 / sH reuse
        {   // stage W2^T chunk: [16 experts][128 k]
            int e  = tid & 15;
            int kb = (tid >> 4) * 8;
            for (int i = 0; i < 8; ++i) {
                int kk = kb + i;
                sW2[e][kk] = W2[(j * 128 + kk) * EE + e];
            }
        }

        v8f acc = {0.f,0.f,0.f,0.f,0.f,0.f,0.f,0.f};
        for (int k0 = 0; k0 < DD; k0 += 64) {
            __syncthreads();
            {   // stage x tile [16][64]: 4 floats per thread
                int idx = tid * 4;
                int r = idx >> 6, c = idx & 63;
                float4 v = *(const float4*)&x[(size_t)(tok0 + r) * DD + k0 + c];
                *(float4*)&sX[r][c] = v;
            }
            {   // stage W1^T tile [128 n][64 k]: 32 per thread
                int n  = tid & 127;
                int kb = (tid >> 7) * 32;
                const float* src = &W1[(size_t)(k0 + kb) * HH + j * 128 + n];
                for (int i = 0; i < 32; ++i)
                    sW1[n][kb + i] = src[(size_t)i * HH];
            }
            __syncthreads();

            const float* aRow = sX[m];
            const float* bRow = sW1[wave * 16 + m];
            for (int kk = 0; kk < 64; kk += 4) {
                v2f a = *(const v2f*)&aRow[kk + 2 * g];   // A[m][k..k+1]
                v2f b = *(const v2f*)&bRow[kk + 2 * g];   // B[k..k+1][n]^T
                acc = __builtin_amdgcn_wmma_f32_16x16x4_f32(
                          false, a, false, b, (short)0, acc, false, false);
            }
        }

        // bias + relu; lane's C columns are h-col (j*128 + wave*16 + m)
        {
            float bias = b1[j * 128 + wave * 16 + m];
            for (int r = 0; r < 8; ++r) {
                float h = acc[r] + bias;
                sH[wave][r + 8 * g][m] = h > 0.f ? h : 0.f;
            }
        }

        // logits += h_tile(16x16) @ W2chunk(16x16): K=16 -> 4 fp32 WMMAs
        for (int kw = 0; kw < 16; kw += 4) {
            v2f a = *(const v2f*)&sH[wave][m][kw + 2 * g];
            v2f b = *(const v2f*)&sW2[m][wave * 16 + kw + 2 * g];
            logit_acc = __builtin_amdgcn_wmma_f32_16x16x4_f32(
                            false, a, false, b, (short)0, logit_acc, false, false);
        }
    }

    // cross-wave logits reduction (C layout: row = r+8g token, col = expert)
    for (int r = 0; r < 8; ++r)
        atomicAdd(&sLogits[r + 8 * g][m], logit_acc[r]);
    __syncthreads();

    // softmax + threshold: 16 lanes, one token each
    if (tid < 16) {
        float l[EE];
        float mx = -3.0e38f;
        for (int e = 0; e < EE; ++e) {
            l[e] = sLogits[tid][e] + b2[e];
            mx = fmaxf(mx, l[e]);
        }
        float s = 0.f;
        for (int e = 0; e < EE; ++e) { l[e] = expf(l[e] - mx); s += l[e]; }
        float inv = 1.0f / s;
        int k = 0;
        for (int e = 0; e < EE; ++e) {
            bool sel = (l[e] >= TAU);        // p >= TAU*pmax
            k += sel ? 1 : 0;
            gates[(size_t)(tok0 + tid) * EE + e] = sel ? l[e] * inv : 0.0f;
        }
        kout[tok0 + tid] = (float)k;
    }
}

// =====================================================================
// Kernel 2: expert compute.  out = sum_e gate_e * (x @ We[e] + be[e]).
// Block = 16 tokens x 128 out-cols, 8 waves (16 cols each).
// x tile converted to bf16 ONCE per block.  We slabs streamed with
// double-buffered GLOBAL_LOAD_ASYNC_TO_LDS_B128 (ASYNCcnt pipelined),
// converted f32->bf16 at register-load time; bf16 WMMA 16x16x32 with
// fp32 accumulate; per-tile expert skip on zero gates.
// =====================================================================
#define XPAD 8          // sXe row stride 2056 u16
#define KC   64         // async K slab
#define BP   132        // sB32 row stride (f32): 132 dw -> halves on disjoint banks

__global__ __launch_bounds__(256) void expert_kernel(
    const float* __restrict__ x,  const float* __restrict__ We,
    const float* __restrict__ be, const float* __restrict__ gates,
    float* __restrict__ out)
{
    __shared__ __align__(16) unsigned short sXe[16][DD + XPAD];   // ~66 KB
#if USE_ASYNC
    __shared__ __align__(16) float sB32[2][KC][BP];               // ~67.6 KB
#else
    __shared__ __align__(16) unsigned short sB[128][136];         // ~34 KB
#endif
    __shared__ __align__(16) float sG[16][16];

    const int tid  = threadIdx.x;
    const int wave = tid >> 5;
    const int lane = tid & 31;
    const int m    = lane & 15;
    const int g    = lane >> 4;
    const int tok0 = blockIdx.x * 16;
    const int n0   = blockIdx.y * 128;
    const int col  = n0 + wave * 16 + m;      // this lane's output column
    const int nloc = wave * 16 + m;           // local column inside the slab

    sG[tid >> 4][tid & 15] = gates[(size_t)(tok0 + (tid >> 4)) * EE + (tid & 15)];

    // stage whole x tile once (f32 -> bf16): 128 elems per thread
    {
        int r  = tid >> 4;
        int cb = (tid & 15) * 128;
        const float* src = &x[(size_t)(tok0 + r) * DD + cb];
        for (int i = 0; i < 128; i += 4) {
            float4 v = *(const float4*)&src[i];
            unsigned int lo = (unsigned int)f2bf(v.x) | ((unsigned int)f2bf(v.y) << 16);
            unsigned int hi = (unsigned int)f2bf(v.z) | ((unsigned int)f2bf(v.w) << 16);
            unsigned int* dst = (unsigned int*)&sXe[r][cb + i];
            dst[0] = lo;
            dst[1] = hi;
        }
    }
    __syncthreads();

    v8f outacc = {0.f,0.f,0.f,0.f,0.f,0.f,0.f,0.f};

    for (int e = 0; e < EE; ++e) {
        // uniform tile-level skip: all threads compute the same verdict
        bool nz = false;
        for (int t = 0; t < 16; ++t) nz |= (sG[t][e] != 0.0f);
        if (!nz) continue;

        v8f acc = {0.f,0.f,0.f,0.f,0.f,0.f,0.f,0.f};

#if USE_ASYNC
        // ---- double-buffered async DMA pipeline over K ----
        // issue slab k0=0 into buf 0: [KC k][128 n] f32, 8 x b128 per thread
        {
            for (int i = 0; i < 8; ++i) {
                int gi = tid + i * 256;
                int k  = gi >> 5, c4 = gi & 31;
                const float* gsrc = &We[((size_t)e * DD + k) * OO + n0 + c4 * 4];
                float* ldst = &sB32[0][k][c4 * 4];
                __builtin_amdgcn_global_load_async_to_lds_b128(
                    (gv4i_p)(v4i*)(float*)gsrc, (lv4i_p)(v4i*)ldst, 0, 0);
            }
        }
        __builtin_amdgcn_s_wait_asynccnt(0);
        __syncthreads();

        for (int k0 = 0; k0 < DD; k0 += KC) {
            const int cur = (k0 / KC) & 1;
            if (k0 + KC < DD) {     // prefetch next slab into the other buffer
                for (int i = 0; i < 8; ++i) {
                    int gi = tid + i * 256;
                    int k  = gi >> 5, c4 = gi & 31;
                    const float* gsrc =
                        &We[((size_t)e * DD + k0 + KC + k) * OO + n0 + c4 * 4];
                    float* ldst = &sB32[cur ^ 1][k][c4 * 4];
                    __builtin_amdgcn_global_load_async_to_lds_b128(
                        (gv4i_p)(v4i*)(float*)gsrc, (lv4i_p)(v4i*)ldst, 0, 0);
                }
            }

            const unsigned short* aRow = &sXe[m][k0];
            const float*          bbuf = &sB32[cur][0][0];
            for (int kc = 0; kc < KC; kc += 32) {
                BfPack ap;
                ap.q[0] = *(const v4u*)&aRow[kc + 8 * g];
                ap.q[1] = *(const v4u*)&aRow[kc + 16 + 8 * g];
                v16bf bv;
#pragma unroll
                for (int v = 0; v < 8; ++v) {
                    int k = kc + ((v < 4) ? (2 * v + 8 * g)
                                          : (16 + 2 * (v - 4) + 8 * g));
                    bv[2 * v]     = (__bf16)bbuf[k * BP + nloc];
                    bv[2 * v + 1] = (__bf16)bbuf[(k + 1) * BP + nloc];
                }
                acc = __builtin_amdgcn_wmma_f32_16x16x32_bf16(
                          false, ap.v, false, bv, (short)0, acc, false, false);
            }

            __syncthreads();        // all reads of buf[cur] done
            if (k0 + KC < DD) {
                __builtin_amdgcn_s_wait_asynccnt(0);   // next slab landed
                __syncthreads();    // async LDS writes visible to all waves
            }
        }
#else
        // ---- synchronous fallback: stage->convert->LDS, 128-wide K slabs ----
        for (int k0 = 0; k0 < DD; k0 += 128) {
            __syncthreads();
            {
                int nn = tid & 127;
                int kb = (tid >> 7) * 64;
                const float* src = &We[((size_t)e * DD + k0 + kb) * OO + n0 + nn];
                for (int i = 0; i < 64; i += 2) {
                    float a0 = src[(size_t)(i    ) * OO];
                    float a1 = src[(size_t)(i + 1) * OO];
                    *(unsigned int*)&sB[nn][kb + i] =
                        (unsigned int)f2bf(a0) | ((unsigned int)f2bf(a1) << 16);
                }
            }
            if (k0 + 128 < DD) {
                int idx = tid * 2;
                for (int p = 0; p < 2; ++p, ++idx) {
                    int row = idx >> 2, seg = idx & 3;
                    __builtin_prefetch(
                        &We[((size_t)e * DD + k0 + 128 + row) * OO + n0 + seg * 32],
                        0, 1);
                }
            }
            __syncthreads();

            const unsigned short* aRow = &sXe[m][k0];
            const unsigned short* bRow = sB[nloc];
            for (int kc = 0; kc < 128; kc += 32) {
                BfPack ap, bp;
                ap.q[0] = *(const v4u*)&aRow[kc + 8 * g];
                ap.q[1] = *(const v4u*)&aRow[kc + 16 + 8 * g];
                bp.q[0] = *(const v4u*)&bRow[kc + 8 * g];
                bp.q[1] = *(const v4u*)&bRow[kc + 16 + 8 * g];
                acc = __builtin_amdgcn_wmma_f32_16x16x32_bf16(
                          false, ap.v, false, bp.v, (short)0, acc, false, false);
            }
        }
#endif

        float bev = be[(size_t)e * OO + col];
        for (int r = 0; r < 8; ++r)
            outacc[r] += sG[r + 8 * g][e] * (acc[r] + bev);
    }

    for (int r = 0; r < 8; ++r)
        out[(size_t)(tok0 + r + 8 * g) * OO + col] = outacc[r];
}

// =====================================================================
extern "C" void kernel_launch(void* const* d_in, const int* in_sizes, int n_in,
                              void* d_out, int out_size, void* d_ws, size_t ws_size,
                              hipStream_t stream) {
    (void)in_sizes; (void)n_in; (void)out_size; (void)ws_size;

    const float* x  = (const float*)d_in[0];
    const float* W1 = (const float*)d_in[1];
    const float* b1 = (const float*)d_in[2];
    const float* W2 = (const float*)d_in[3];
    const float* b2 = (const float*)d_in[4];
    const float* We = (const float*)d_in[5];
    const float* be = (const float*)d_in[6];

    float* out   = (float*)d_out;
    float* kout  = out + (size_t)BT * OO;   // k_per_token appended flat
    float* gates = (float*)d_ws;            // [BT][EE] = 512 KB scratch

    router_kernel<<<dim3(BT / 16), 256, 0, stream>>>(x, W1, b1, W2, b2, gates, kout);
    expert_kernel<<<dim3(BT / 16, OO / 128), 256, 0, stream>>>(x, We, be, gates, out);
}
